// LSTMAddonsDecoder_31937376813722
// MI455X (gfx1250) — matrix operand order repaired and checked by reading, not compiled, not run on hardware
//
#include <hip/hip_runtime.h>
#include <hip/hip_bf16.h>

// ---------------------------------------------------------------------------
// LSTM forward on MI455X (gfx1250, wave32, WMMA).
//   z = [x_t | h_{t-1}] @ [W;U] + b   (K = 128+512 = 640, bf16 WMMA, f32 acc)
//   gates -> c,h update (fp32, c kept in registers for all 512 steps)
//   out = hs @ Wo + bo                (bf16 WMMA, f32 acc)
// Recurrence runs as ONE persistent kernel (64 blocks) with a device-wide
// monotonic-counter barrier per timestep; h is double-buffered in global mem.
// ---------------------------------------------------------------------------

typedef __bf16 bf16_t;
typedef __attribute__((ext_vector_type(16))) __bf16 v16bf;
typedef __attribute__((ext_vector_type(8)))  __bf16 v8bf;
typedef __attribute__((ext_vector_type(8)))  float  v8f;

union AFrag { v16bf v; v8bf h[2]; };

#define WMMA_BF16(a, b, c) \
  __builtin_amdgcn_wmma_f32_16x16x32_bf16(false, (a), false, (b), (short)0, (c), false, false)

__device__ __forceinline__ float sigmoid_f(float x) {
  return 1.0f / (1.0f + __expf(-x));
}
__device__ __forceinline__ float tanh_f(float x) {
  // 1 - 2/(e^{2x}+1): saturates correctly at +/-1 without inf/inf NaNs
  float e = __expf(2.0f * x);
  return 1.0f - 2.0f / (e + 1.0f);
}

// --------------------------- prep kernels ----------------------------------

// Pack [W;U] (fp32, row-major [K][2048]) into B-fragment layout:
//   dst[((ntile*20 + kc)*32 + lane)*16 + e] = WU[kc*32 + (e>>3)*16 + (lane>>4)*8 + (e&7)][ntile*16 + (lane&15)]
__global__ void pack_wu_kernel(const float* __restrict__ W, const float* __restrict__ U,
                               bf16_t* __restrict__ dst, int n) {
  int idx = blockIdx.x * blockDim.x + threadIdx.x;
  if (idx >= n) return;
  int e = idx & 15;
  int t = idx >> 4;
  int lane = t & 31; t >>= 5;
  int kc = t % 20;
  int nt = t / 20;
  int ncol = nt * 16 + (lane & 15);
  int k = kc * 32 + (e >> 3) * 16 + (lane >> 4) * 8 + (e & 7);          // 0..639
  float v = (k < 128) ? W[k * 2048 + ncol] : U[(k - 128) * 2048 + ncol];
  dst[idx] = (bf16_t)v;
}

// Pack Wo (fp32 [512][256]) into B-fragment layout, 16 n-tiles x 16 k-chunks.
__global__ void pack_wo_kernel(const float* __restrict__ Wo, bf16_t* __restrict__ dst, int n) {
  int idx = blockIdx.x * blockDim.x + threadIdx.x;
  if (idx >= n) return;
  int e = idx & 15;
  int t = idx >> 4;
  int lane = t & 31; t >>= 5;
  int kc = t & 15;
  int nt = t >> 4;
  int ncol = nt * 16 + (lane & 15);
  int k = kc * 32 + (e >> 3) * 16 + (lane >> 4) * 8 + (e & 7);          // 0..511
  dst[idx] = (bf16_t)Wo[k * 256 + ncol];
}

// fp32 -> bf16 elementwise (x stays [B][T][D]).
__global__ void cvt_x_kernel(const float* __restrict__ src, bf16_t* __restrict__ dst, int n) {
  int idx = blockIdx.x * blockDim.x + threadIdx.x;
  if (idx < n) dst[idx] = (bf16_t)src[idx];
}

// h0 -> h_bf buffer 0; zero the global barrier counter.
__global__ void init_h_kernel(const float* __restrict__ h0, bf16_t* __restrict__ h_bf,
                              unsigned* __restrict__ cnt, int n) {
  int idx = blockIdx.x * blockDim.x + threadIdx.x;
  if (idx == 0) *cnt = 0u;
  if (idx < n) h_bf[idx] = (bf16_t)h0[idx];
}

// --------------------------- recurrent kernel ------------------------------
// 64 blocks x 256 threads (8 waves). Each block owns 2 output tile-pairs;
// each pair = 4 waves, one per gate. Per wave per step: 20 wmma (K=640).
__global__ __launch_bounds__(256) void lstm_recurrent_kernel(
    const bf16_t* __restrict__ x_bf,   // [64][512][128]
    const bf16_t* __restrict__ wu,     // packed [128 ntiles][20 kc][32 lane][16]
    const float*  __restrict__ bias,   // [2048]
    const float*  __restrict__ c0,     // [64][512]
    bf16_t* __restrict__ h_bf,         // [2][64][512] double buffer
    bf16_t* __restrict__ hs,           // [64][512][512]
    unsigned* __restrict__ cnt,
    int nblocks) {
  __shared__ float zbuf[2][4][16][16];

  const int tid   = threadIdx.x;
  const int wave  = tid >> 5;
  const int lane  = tid & 31;
  const int pairL = wave >> 2;          // 0..1
  const int gate  = wave & 3;           // 0=i 1=f 2=g 3=o
  const int pair  = blockIdx.x * 2 + pairL;   // 0..127
  const int mtile = pair >> 5;          // 0..3   (batch rows /16)
  const int jtile = pair & 31;          // 0..31  (hidden cols /16)
  const int ntile = gate * 32 + jtile;  // column tile in 4H space

  const int laneRow  = lane & 15;       // A-matrix row within tile
  const int half     = lane >> 4;       // K-half selector
  const int rowA     = mtile * 16 + laneRow;
  const int colD     = lane & 15;       // D/C-matrix column
  const int rowDbase = half * 8;        // D rows rowDbase..rowDbase+7
  const int jcol     = jtile * 16 + colD;

  // Persistent cell state tile (used by gate-0 waves) + per-column biases.
  float creg[8];
  float biasv[4];
#pragma unroll
  for (int r = 0; r < 8; ++r)
    creg[r] = c0[(mtile * 16 + rowDbase + r) * 512 + jcol];
#pragma unroll
  for (int g = 0; g < 4; ++g)
    biasv[g] = bias[g * 512 + jcol];

  const bf16_t* bbase = wu + ((size_t)ntile * 20) * 32 * 16;

  for (int t = 0; t < 512; ++t) {
    const int rb = t & 1;               // read buffer; write rb^1
    v8f acc = {};
    const bf16_t* xrow = x_bf + ((size_t)rowA * 512 + t) * 128 + half * 8;
    const bf16_t* hrow = h_bf + (size_t)rb * 32768 + (size_t)rowA * 512 + half * 8;

#pragma unroll
    for (int kc = 0; kc < 20; ++kc) {
      const bf16_t* src = (kc < 4) ? (xrow + kc * 32) : (hrow + (kc - 4) * 32);
      AFrag a;
      a.h[0] = *(const v8bf*)(src);
      a.h[1] = *(const v8bf*)(src + 16);
      v16bf b = *(const v16bf*)(bbase + ((size_t)kc * 32 + lane) * 16);
      acc = WMMA_BF16(a.v, b, acc);
    }

    // Exchange gate z-tiles through LDS.
#pragma unroll
    for (int r = 0; r < 8; ++r)
      zbuf[pairL][gate][rowDbase + r][colD] = acc[r];
    __syncthreads();

    if (gate == 0) {
#pragma unroll
      for (int r = 0; r < 8; ++r) {
        const int rr = rowDbase + r;
        float zi = zbuf[pairL][0][rr][colD] + biasv[0];
        float zf = zbuf[pairL][1][rr][colD] + biasv[1];
        float zg = zbuf[pairL][2][rr][colD] + biasv[2];
        float zo = zbuf[pairL][3][rr][colD] + biasv[3];
        float iv = sigmoid_f(zi);
        float fv = sigmoid_f(zf);
        float gv = tanh_f(zg);
        float ov = sigmoid_f(zo);
        float cn = fv * creg[r] + iv * gv;
        creg[r] = cn;
        float hn = ov * tanh_f(cn);
        const int brow = mtile * 16 + rr;            // batch index
        bf16_t hb = (bf16_t)hn;
        h_bf[(size_t)(rb ^ 1) * 32768 + (size_t)brow * 512 + jcol] = hb;
        hs[((size_t)brow * 512 + t) * 512 + jcol] = hb;
      }
    }

    // Device-wide barrier (monotonic counter, no reset -> no sense races).
    __threadfence();
    __syncthreads();
    if (tid == 0) {
      __hip_atomic_fetch_add(cnt, 1u, __ATOMIC_RELEASE, __HIP_MEMORY_SCOPE_AGENT);
      const unsigned target = (unsigned)nblocks * (unsigned)(t + 1);
      while (__hip_atomic_load(cnt, __ATOMIC_ACQUIRE, __HIP_MEMORY_SCOPE_AGENT) < target) {
        __builtin_amdgcn_s_sleep(2);
      }
    }
    __syncthreads();
    __threadfence();   // acquire: make new h_bf visible to every wave's cache path
  }
}

// --------------------------- output GEMM -----------------------------------
// out[32768 x 256] = hs[32768 x 512] @ Wo + bo. One 16x16 tile per wave.
__global__ __launch_bounds__(256) void out_gemm_kernel(
    const bf16_t* __restrict__ hs, const bf16_t* __restrict__ wo,
    const float* __restrict__ bo, float* __restrict__ out) {
  const int tid  = threadIdx.x;
  const int wave = tid >> 5;
  const int lane = tid & 31;
  const int tile = blockIdx.x * 8 + wave;     // 0..32767
  const int mt = tile >> 4;                   // 0..2047
  const int nt = tile & 15;                   // 0..15
  const int laneRow = lane & 15;
  const int half = lane >> 4;
  const int row = mt * 16 + laneRow;

  v8f acc = {};
  const bf16_t* arow = hs + (size_t)row * 512 + half * 8;
  const bf16_t* bbase = wo + ((size_t)nt * 16) * 32 * 16;
#pragma unroll
  for (int kc = 0; kc < 16; ++kc) {
    AFrag a;
    const bf16_t* src = arow + kc * 32;
    a.h[0] = *(const v8bf*)(src);
    a.h[1] = *(const v8bf*)(src + 16);
    v16bf b = *(const v16bf*)(bbase + ((size_t)kc * 32 + lane) * 16);
    acc = WMMA_BF16(a.v, b, acc);
  }
  const int colD = lane & 15;
  const int ncol = nt * 16 + colD;
  const float bv = bo[ncol];
#pragma unroll
  for (int r = 0; r < 8; ++r) {
    const int rg = mt * 16 + half * 8 + r;
    out[(size_t)rg * 256 + ncol] = acc[r] + bv;
  }
}

// --------------------------- launch ----------------------------------------

extern "C" void kernel_launch(void* const* d_in, const int* in_sizes, int n_in,
                              void* d_out, int out_size, void* d_ws, size_t ws_size,
                              hipStream_t stream) {
  (void)in_sizes; (void)n_in; (void)out_size; (void)ws_size;
  const float* x  = (const float*)d_in[0];  // [64][512][128]
  const float* h0 = (const float*)d_in[1];  // [64][512]
  const float* c0 = (const float*)d_in[2];  // [64][512]
  const float* W  = (const float*)d_in[3];  // [128][2048]
  const float* U  = (const float*)d_in[4];  // [512][2048]
  const float* b  = (const float*)d_in[5];  // [2048]
  const float* Wo = (const float*)d_in[6];  // [512][256]
  const float* bo = (const float*)d_in[7];  // [256]

  char* ws = (char*)d_ws;
  bf16_t*  WUpk = (bf16_t*)(ws + 0);          // 1,310,720 el = 2,621,440 B
  bf16_t*  Wopk = (bf16_t*)(ws + 2621440);    //   131,072 el =   262,144 B
  bf16_t*  x_bf = (bf16_t*)(ws + 2883584);    // 4,194,304 el = 8,388,608 B
  bf16_t*  h_bf = (bf16_t*)(ws + 11272192);   // 2*32,768 el  =   131,072 B
  bf16_t*  hs   = (bf16_t*)(ws + 11403264);   // 16,777,216 el = 33,554,432 B
  unsigned* cnt = (unsigned*)(ws + 44957696); // barrier counter

  pack_wu_kernel<<<(1310720 + 255) / 256, 256, 0, stream>>>(W, U, WUpk, 1310720);
  pack_wo_kernel<<<(131072 + 255) / 256, 256, 0, stream>>>(Wo, Wopk, 131072);
  cvt_x_kernel<<<(4194304 + 255) / 256, 256, 0, stream>>>(x, x_bf, 4194304);
  init_h_kernel<<<(32768 + 255) / 256, 256, 0, stream>>>(h0, h_bf, cnt, 32768);

  lstm_recurrent_kernel<<<64, 256, 0, stream>>>(x_bf, WUpk, b, c0, h_bf, hs, cnt, 64);

  out_gemm_kernel<<<4096, 256, 0, stream>>>(hs, Wopk, bo, (float*)d_out);
}
